// DecoderLayer_50697793962249
// MI455X (gfx1250) — matrix operand order repaired and checked by reading, not compiled, not run on hardware
//
#include <hip/hip_runtime.h>
#include <math.h>

// ---------------- problem constants ----------------
#define D_MODEL 768
#define HEADS   12
#define D_HEAD  64
#define BATCH   16
#define SPR     256
#define SIM     1024
#define LN_EPS  1e-5f

typedef __attribute__((ext_vector_type(16))) __bf16 bf16x16;
typedef __attribute__((ext_vector_type(8)))  float  f32x8;
typedef unsigned short u16;

union BF16Frag { bf16x16 v; u16 s[16]; };

// round-to-nearest-even f32 -> bf16 bit pattern
__device__ __forceinline__ u16 f2bf(float f) {
  union { float f; unsigned u; } x; x.f = f;
  unsigned u = x.u;
  return (u16)((u + 0x7fffu + ((u >> 16) & 1u)) >> 16);
}
__device__ __forceinline__ float bf2f(u16 h) {
  union { unsigned u; float f; } x; x.u = ((unsigned)h) << 16; return x.f;
}

// LDS byte offset of a __shared__ object (low 32 bits of its flat address)
__device__ __forceinline__ unsigned ldsOff(const void* p) {
  return (unsigned)(unsigned long long)p;
}
// CDNA5 async copy: global -> LDS, 16 bytes per lane, tracked by ASYNCcnt
__device__ __forceinline__ void asyncLoadB128(unsigned lds, const u16* g) {
  asm volatile("global_load_async_to_lds_b128 %0, %1, off"
               :: "v"(lds), "v"(g) : "memory");
}
__device__ __forceinline__ void asyncWait() {
  asm volatile("s_wait_asynccnt 0x0" ::: "memory");
}

// ==================================================================
// Weight prep: W f32 [K=768][N=768] -> Wt bf16 [N][K] (transpose+convert)
// so GEMM B-fragments are K-contiguous in LDS.
// ==================================================================
__global__ __launch_bounds__(256)
void wt_convert_kernel(const float* __restrict__ W, u16* __restrict__ Wt)
{
  __shared__ float t[32][33];
  const int tx = threadIdx.x & 31;
  const int ty = threadIdx.x >> 5;       // 0..7
  const int nB = blockIdx.x * 32;
  const int kB = blockIdx.y * 32;
#pragma unroll
  for (int i = 0; i < 32; i += 8)
    t[ty + i][tx] = W[(size_t)(kB + ty + i) * D_MODEL + nB + tx];
  __syncthreads();
#pragma unroll
  for (int i = 0; i < 32; i += 8)
    Wt[(size_t)(nB + ty + i) * D_MODEL + kB + tx] = f2bf(t[tx][ty + i]);
}

// ==================================================================
// Fused residual-add + LayerNorm. One wave per row of 768.
// sum_out (optional) is f32; ln_out is bf16 (feeds GEMM A operands).
// ==================================================================
__global__ __launch_bounds__(256)
void add_ln_kernel(const float* __restrict__ a, const float* __restrict__ b,
                   float* __restrict__ sum_out, u16* __restrict__ ln_out,
                   const float* __restrict__ gamma, const float* __restrict__ beta,
                   int rows)
{
  const int wid  = threadIdx.x >> 5;
  const int lane = threadIdx.x & 31;
  const int row  = blockIdx.x * 8 + wid;
  if (row >= rows) return;
  const size_t base = (size_t)row * D_MODEL;

  float s[24];
  float sum = 0.f, sum2 = 0.f;
#pragma unroll
  for (int i = 0; i < 24; ++i) {
    int c = lane + i * 32;
    float v = a[base + c] + b[base + c];
    s[i] = v; sum += v; sum2 += v * v;
  }
#pragma unroll
  for (int m = 1; m < 32; m <<= 1) {
    sum  += __shfl_xor(sum,  m, 32);
    sum2 += __shfl_xor(sum2, m, 32);
  }
  const float mean = sum * (1.0f / D_MODEL);
  const float var  = sum2 * (1.0f / D_MODEL) - mean * mean;
  const float inv  = rsqrtf(var + LN_EPS);
#pragma unroll
  for (int i = 0; i < 24; ++i) {
    int c = lane + i * 32;
    if (sum_out) sum_out[base + c] = s[i];
    ln_out[base + c] = f2bf((s[i] - mean) * inv * gamma[c] + beta[c]);
  }
}

// ==================================================================
// GEMM: acc = A[M x 768](bf16) @ Wt^T (Wt bf16 [768 x 768], N-major)
// epilogue: + bias (+ f32 resid) (+ relu); store f32 (Cf) and/or
// bf16 (Cb; vtSk>0 -> per-head-transposed V layout [B,H,64,Sk]).
// Block tile 128x64, 8 waves 4x2, 2x2 WMMA frags per wave, async LDS staging.
// ==================================================================
#define BM 128
#define BN 64
#define BK 32
#define LDA_S 40
#define LDB_S 40

__global__ __launch_bounds__(256)
void gemm_bf16_kernel(const u16* __restrict__ A, const u16* __restrict__ Wt,
                      const float* __restrict__ bias, const float* __restrict__ resid,
                      float* __restrict__ Cf, u16* __restrict__ Cb,
                      int M, int relu, int vtSk)
{
  __shared__ u16 sA[BM * LDA_S];
  __shared__ u16 sB[BN * LDB_S];

  const int tid   = threadIdx.x;
  const int lane  = tid & 31;
  const int wid   = tid >> 5;
  const int half  = lane >> 4;
  const int l15   = lane & 15;
  const int waveM = wid >> 1;
  const int waveN = wid & 1;
  const int rowBlk = blockIdx.x * BM;
  const int colBlk = blockIdx.y * BN;

  f32x8 acc[2][2];
#pragma unroll
  for (int i = 0; i < 2; ++i)
#pragma unroll
    for (int j = 0; j < 2; ++j)
#pragma unroll
      for (int r = 0; r < 8; ++r) acc[i][j][r] = 0.f;

  for (int kb = 0; kb < D_MODEL; kb += BK) {
    // async-stage A tile 128x32 bf16 (2 x b128 per thread)
#pragma unroll
    for (int i = 0; i < 2; ++i) {
      int idx = i * 256 + tid;
      int r = idx >> 2, ch = (idx & 3) * 8;
      asyncLoadB128(ldsOff(&sA[r * LDA_S + ch]),
                    A + (size_t)(rowBlk + r) * D_MODEL + kb + ch);
    }
    // async-stage Wt tile 64x32 bf16 (1 x b128 per thread)
    {
      int r = tid >> 2, ch = (tid & 3) * 8;
      asyncLoadB128(ldsOff(&sB[r * LDB_S + ch]),
                    Wt + (size_t)(colBlk + r) * D_MODEL + kb + ch);
    }
    if (kb + BK < D_MODEL)
      __builtin_prefetch(A + (size_t)(rowBlk + (tid >> 1)) * D_MODEL + kb + BK, 0, 0);
    asyncWait();
    __syncthreads();

    BF16Frag af[2], bfr[2];
#pragma unroll
    for (int i = 0; i < 2; ++i) {
      const int mrow = waveM * 32 + i * 16 + l15;
#pragma unroll
      for (int v = 0; v < 8; ++v) {
        // A layout: lanes 0-15 K={0..15}; lanes 16-31 K={16..31}
        int kp = (v << 1) + (half << 4);
        af[i].s[2 * v]     = sA[mrow * LDA_S + kp];
        af[i].s[2 * v + 1] = sA[mrow * LDA_S + kp + 1];
      }
    }
#pragma unroll
    for (int j = 0; j < 2; ++j) {
      const int ncol = waveN * 32 + j * 16 + l15;
#pragma unroll
      for (int e = 0; e < 16; ++e)  // K-contiguous thanks to transposed staging
        bfr[j].s[e] = sB[ncol * LDB_S + half * 16 + e];
    }
#pragma unroll
    for (int i = 0; i < 2; ++i)
#pragma unroll
      for (int j = 0; j < 2; ++j)
        acc[i][j] = __builtin_amdgcn_wmma_f32_16x16x32_bf16(
            false, af[i].v, false, bfr[j].v, (short)0, acc[i][j], false, false);
    __syncthreads();
  }

  // epilogue folded into acc
#pragma unroll
  for (int i = 0; i < 2; ++i)
#pragma unroll
    for (int j = 0; j < 2; ++j)
#pragma unroll
      for (int r = 0; r < 8; ++r) {
        int row = rowBlk + waveM * 32 + i * 16 + half * 8 + r;
        int col = colBlk + waveN * 32 + j * 16 + l15;
        float v = acc[i][j][r] + bias[col];
        if (resid) v += resid[(size_t)row * D_MODEL + col];
        if (relu)  v = fmaxf(v, 0.0f);
        acc[i][j][r] = v;
      }

  if (Cf) {
#pragma unroll
    for (int i = 0; i < 2; ++i)
#pragma unroll
      for (int j = 0; j < 2; ++j)
#pragma unroll
        for (int r = 0; r < 8; ++r) {
          int row = rowBlk + waveM * 32 + i * 16 + half * 8 + r;
          int col = colBlk + waveN * 32 + j * 16 + l15;
          Cf[(size_t)row * D_MODEL + col] = acc[i][j][r];
        }
  }
  if (Cb) {
#pragma unroll
    for (int i = 0; i < 2; ++i)
#pragma unroll
      for (int j = 0; j < 2; ++j)
#pragma unroll
        for (int r = 0; r < 8; ++r) {
          int row = rowBlk + waveM * 32 + i * 16 + half * 8 + r;
          int col = colBlk + waveN * 32 + j * 16 + l15;
          if (vtSk > 0) { // per-head transposed V: [B, H, 64, Sk]
            int b = row / vtSk, kv = row - b * vtSk;
            int h = col >> 6, d = col & 63;
            Cb[(((size_t)b * HEADS + h) * D_HEAD + d) * vtSk + kv] = f2bf(acc[i][j][r]);
          } else {
            Cb[(size_t)row * D_MODEL + col] = f2bf(acc[i][j][r]);
          }
        }
  }
}

// ==================================================================
// Fused flash attention (bf16 operands, f32 softmax/accum).
// grid.x = B*HEADS, grid.y = Sq/128; each wave owns 16 q-rows.
// Q,K: bf16 [B*S, 768]; Vt: bf16 [B, H, 64, Sk]; O: bf16 [B*Sq, 768].
// ==================================================================
__global__ __launch_bounds__(256)
void attn_kernel(const u16* __restrict__ Q, const u16* __restrict__ K,
                 const u16* __restrict__ Vt, u16* __restrict__ O,
                 int Sq, int Sk, float scale)
{
  __shared__ u16 sK[32 * 72];     // [kv][d]
  __shared__ u16 sV[64 * LDB_S];  // [d][kv] (already transposed in HBM)
  __shared__ float sP[8][16][34];

  const int tid  = threadIdx.x;
  const int lane = tid & 31;
  const int wid  = tid >> 5;
  const int half = lane >> 4;
  const int l15  = lane & 15;
  const int b = blockIdx.x / HEADS;
  const int h = blockIdx.x % HEADS;
  const int q0 = blockIdx.y * 128 + wid * 16;
  const int hc = h * D_HEAD;
  const size_t qrow = (size_t)b * Sq + q0;
  const size_t vtbase = ((size_t)b * HEADS + h) * D_HEAD;

  // Q fragments, loaded once directly from bf16 HBM
  BF16Frag qa[2];
#pragma unroll
  for (int kk = 0; kk < 2; ++kk) {
    const u16* qp = Q + (qrow + l15) * D_MODEL + hc + kk * 32;
#pragma unroll
    for (int v = 0; v < 8; ++v) {
      int kp = (v << 1) + (half << 4);
      qa[kk].s[2 * v]     = qp[kp];
      qa[kk].s[2 * v + 1] = qp[kp + 1];
    }
  }

  float mrow[8], lrow[8];
  f32x8 oacc[4];
#pragma unroll
  for (int r = 0; r < 8; ++r) { mrow[r] = -1e30f; lrow[r] = 0.f; }
#pragma unroll
  for (int n = 0; n < 4; ++n)
#pragma unroll
    for (int r = 0; r < 8; ++r) oacc[n][r] = 0.f;

  for (int j0 = 0; j0 < Sk; j0 += 32) {
    __syncthreads();
    { // async-stage K tile 32x64 (1 x b128 per thread)
      int kr = tid >> 3, ch = (tid & 7) * 8;
      asyncLoadB128(ldsOff(&sK[kr * 72 + ch]),
                    K + ((size_t)b * Sk + j0 + kr) * D_MODEL + hc + ch);
    }
    { // async-stage Vt tile 64x32 (1 x b128 per thread)
      int d = tid >> 2, ch = (tid & 3) * 8;
      asyncLoadB128(ldsOff(&sV[d * LDB_S + ch]),
                    Vt + (vtbase + d) * Sk + j0 + ch);
    }
    asyncWait();
    __syncthreads();

    // S[16 x 32] = Q @ K^T  (scale applied post-WMMA in f32)
    f32x8 sacc[2];
#pragma unroll
    for (int c = 0; c < 2; ++c)
#pragma unroll
      for (int r = 0; r < 8; ++r) sacc[c][r] = 0.f;
#pragma unroll
    for (int c = 0; c < 2; ++c) {
#pragma unroll
      for (int kk = 0; kk < 2; ++kk) {
        BF16Frag kbf;
#pragma unroll
        for (int e = 0; e < 16; ++e)
          kbf.s[e] = sK[(c * 16 + l15) * 72 + kk * 32 + half * 16 + e];
        sacc[c] = __builtin_amdgcn_wmma_f32_16x16x32_bf16(
            false, qa[kk].v, false, kbf.v, (short)0, sacc[c], false, false);
      }
    }

    // online softmax: a row lives across a 16-lane half in VGPR r
#pragma unroll
    for (int r = 0; r < 8; ++r) {
      float sa = sacc[0][r] * scale, sb = sacc[1][r] * scale;
      float mx = fmaxf(sa, sb);
#pragma unroll
      for (int m = 1; m < 16; m <<= 1) mx = fmaxf(mx, __shfl_xor(mx, m, 32));
      float mn   = fmaxf(mrow[r], mx);
      float corr = __expf(mrow[r] - mn);
      float pa = __expf(sa - mn);
      float pb = __expf(sb - mn);
      float rs = pa + pb;
#pragma unroll
      for (int m = 1; m < 16; m <<= 1) rs += __shfl_xor(rs, m, 32);
      lrow[r] = lrow[r] * corr + rs;
      mrow[r] = mn;
#pragma unroll
      for (int n = 0; n < 4; ++n) oacc[n][r] *= corr;
      int prow = half * 8 + r;
      sP[wid][prow][l15]      = pa;
      sP[wid][prow][16 + l15] = pb;
    }
    __syncthreads();

    // reload P as A-fragment (cross-lane relayout via LDS)
    BF16Frag pf;
#pragma unroll
    for (int v = 0; v < 8; ++v) {
      int kp = (v << 1) + (half << 4);
      pf.s[2 * v]     = f2bf(sP[wid][l15][kp]);
      pf.s[2 * v + 1] = f2bf(sP[wid][l15][kp + 1]);
    }
    // O[16 x 64] += P[16 x 32] @ V[32 x 64]  (V kv-contiguous per d-row)
#pragma unroll
    for (int n = 0; n < 4; ++n) {
      BF16Frag vbf;
#pragma unroll
      for (int e = 0; e < 16; ++e)
        vbf.s[e] = sV[(n * 16 + l15) * LDB_S + half * 16 + e];
      oacc[n] = __builtin_amdgcn_wmma_f32_16x16x32_bf16(
          false, pf.v, false, vbf.v, (short)0, oacc[n], false, false);
    }
  }

  // normalize and write bf16
#pragma unroll
  for (int r = 0; r < 8; ++r) {
    float inv = 1.0f / lrow[r];
    int row = q0 + half * 8 + r;
#pragma unroll
    for (int n = 0; n < 4; ++n)
      O[((size_t)b * Sq + row) * D_MODEL + hc + n * 16 + l15] = f2bf(oacc[n][r] * inv);
  }
}

// ==================================================================
// Host orchestration
// ==================================================================
extern "C" void kernel_launch(void* const* d_in, const int* in_sizes, int n_in,
                              void* d_out, int out_size, void* d_ws, size_t ws_size,
                              hipStream_t stream)
{
  (void)in_sizes; (void)n_in; (void)out_size; (void)ws_size;

  const float* image  = (const float*)d_in[0];
  const float* prompt = (const float*)d_in[1];
  const float* posi   = (const float*)d_in[2];
  const float* posp   = (const float*)d_in[3];
  const float* ln_p1_g = (const float*)d_in[4];
  const float* ln_p1_b = (const float*)d_in[5];
  const float* ln_p2_g = (const float*)d_in[6];
  const float* ln_p2_b = (const float*)d_in[7];
  const float* ln_p3_g = (const float*)d_in[8];
  const float* ln_p3_b = (const float*)d_in[9];
  const float* ln_i1_g = (const float*)d_in[10];
  const float* ln_i1_b = (const float*)d_in[11];
  const float* pp_wq = (const float*)d_in[12]; const float* pp_bq = (const float*)d_in[13];
  const float* pp_wk = (const float*)d_in[14]; const float* pp_bk = (const float*)d_in[15];
  const float* pp_wv = (const float*)d_in[16]; const float* pp_bv = (const float*)d_in[17];
  const float* pp_wo = (const float*)d_in[18]; const float* pp_bo = (const float*)d_in[19];
  const float* pi_wq = (const float*)d_in[20]; const float* pi_bq = (const float*)d_in[21];
  const float* pi_wk = (const float*)d_in[22]; const float* pi_bk = (const float*)d_in[23];
  const float* pi_wv = (const float*)d_in[24]; const float* pi_bv = (const float*)d_in[25];
  const float* pi_wo = (const float*)d_in[26]; const float* pi_bo = (const float*)d_in[27];
  const float* ff_w1 = (const float*)d_in[28]; const float* ff_b1 = (const float*)d_in[29];
  const float* ff_w2 = (const float*)d_in[30]; const float* ff_b2 = (const float*)d_in[31];
  float* out = (float*)d_out;

  const int Mp = BATCH * SPR;    // 4096
  const int Mi = BATCH * SIM;    // 16384
  const size_t U   = (size_t)Mp * D_MODEL;       // 3.15M elems
  const size_t WSZ = (size_t)D_MODEL * D_MODEL;  // 589824

  // workspace: [10 x Wt bf16][2 x f32 U][15 x bf16 U]
  u16* wt = (u16*)d_ws;
  u16* wt_ppq = wt + 0 * WSZ; u16* wt_ppk = wt + 1 * WSZ;
  u16* wt_ppv = wt + 2 * WSZ; u16* wt_ppo = wt + 3 * WSZ;
  u16* wt_piq = wt + 4 * WSZ; u16* wt_pik = wt + 5 * WSZ;
  u16* wt_piv = wt + 6 * WSZ; u16* wt_pio = wt + 7 * WSZ;
  u16* wt_ff1 = wt + 8 * WSZ; u16* wt_ff2 = wt + 9 * WSZ;
  float* fbase = (float*)(wt + 10 * WSZ);
  float* b_p0 = fbase;          // prompt + posp        (U f32)
  float* b_p  = fbase + U;      // residual stream      (U f32)
  u16* ub   = (u16*)(fbase + 2 * U);
  u16* b_x  = ub;               // LN out               (U)
  u16* b_xi = ub + U;           // LN(image+posi)       (4U)
  u16* b_q  = ub + 5 * U;       // q proj / ffn hidden  (U)
  u16* b_k  = ub + 6 * U;       // k proj               (4U)
  u16* b_vt = ub + 10 * U;      // v proj, transposed   (4U)
  u16* b_o  = ub + 14 * U;      // attention out        (U)

  const dim3 blk(256);
  const dim3 gwt(D_MODEL / 32, D_MODEL / 32);
  const dim3 gln(Mp / 8), glni(Mi / 8);
  const dim3 gg(Mp / BM, D_MODEL / BN), ggi(Mi / BM, D_MODEL / BN);
  const dim3 gat(BATCH * HEADS, SPR / 128);
  const float scale = 0.125f;    // 1/sqrt(64)

  // --- weight transpose+convert (once per call) ---
  wt_convert_kernel<<<gwt, blk, 0, stream>>>(pp_wq, wt_ppq);
  wt_convert_kernel<<<gwt, blk, 0, stream>>>(pp_wk, wt_ppk);
  wt_convert_kernel<<<gwt, blk, 0, stream>>>(pp_wv, wt_ppv);
  wt_convert_kernel<<<gwt, blk, 0, stream>>>(pp_wo, wt_ppo);
  wt_convert_kernel<<<gwt, blk, 0, stream>>>(pi_wq, wt_piq);
  wt_convert_kernel<<<gwt, blk, 0, stream>>>(pi_wk, wt_pik);
  wt_convert_kernel<<<gwt, blk, 0, stream>>>(pi_wv, wt_piv);
  wt_convert_kernel<<<gwt, blk, 0, stream>>>(pi_wo, wt_pio);
  wt_convert_kernel<<<gwt, blk, 0, stream>>>(ff_w1, wt_ff1);
  wt_convert_kernel<<<gwt, blk, 0, stream>>>(ff_w2, wt_ff2);

  // --- self-attention block ---
  add_ln_kernel<<<gln, blk, 0, stream>>>(prompt, posp, b_p0, b_x, ln_p1_g, ln_p1_b, Mp);
  gemm_bf16_kernel<<<gg, blk, 0, stream>>>(b_x, wt_ppq, pp_bq, nullptr, nullptr, b_q,  Mp, 0, 0);
  gemm_bf16_kernel<<<gg, blk, 0, stream>>>(b_x, wt_ppk, pp_bk, nullptr, nullptr, b_k,  Mp, 0, 0);
  gemm_bf16_kernel<<<gg, blk, 0, stream>>>(b_x, wt_ppv, pp_bv, nullptr, nullptr, b_vt, Mp, 0, SPR);
  attn_kernel<<<gat, blk, 0, stream>>>(b_q, b_k, b_vt, b_o, SPR, SPR, scale);
  gemm_bf16_kernel<<<gg, blk, 0, stream>>>(b_o, wt_ppo, pp_bo, prompt, b_p, nullptr, Mp, 0, 0);

  // --- cross-attention block ---
  add_ln_kernel<<<gln,  blk, 0, stream>>>(b_p, b_p0, nullptr, b_x, ln_p2_g, ln_p2_b, Mp);
  add_ln_kernel<<<glni, blk, 0, stream>>>(image, posi, nullptr, b_xi, ln_i1_g, ln_i1_b, Mi);
  gemm_bf16_kernel<<<gg,  blk, 0, stream>>>(b_x,  wt_piq, pi_bq, nullptr, nullptr, b_q,  Mp, 0, 0);
  gemm_bf16_kernel<<<ggi, blk, 0, stream>>>(b_xi, wt_pik, pi_bk, nullptr, nullptr, b_k,  Mi, 0, 0);
  gemm_bf16_kernel<<<ggi, blk, 0, stream>>>(b_xi, wt_piv, pi_bv, nullptr, nullptr, b_vt, Mi, 0, SIM);
  attn_kernel<<<gat, blk, 0, stream>>>(b_q, b_k, b_vt, b_o, SPR, SIM, scale);
  gemm_bf16_kernel<<<gg, blk, 0, stream>>>(b_o, wt_pio, pi_bo, b_p, b_p, nullptr, Mp, 0, 0);

  // --- FFN ---
  add_ln_kernel<<<gln, blk, 0, stream>>>(b_p, b_p0, nullptr, b_x, ln_p3_g, ln_p3_b, Mp);
  gemm_bf16_kernel<<<gg, blk, 0, stream>>>(b_x, wt_ff1, ff_b1, nullptr, nullptr, b_q, Mp, 1, 0);
  gemm_bf16_kernel<<<gg, blk, 0, stream>>>(b_q, wt_ff2, ff_b2, nullptr, out, nullptr, Mp, 0, 0);
}
